// HierarchicalContrast_79748952752294
// MI455X (gfx1250) — compile-verified, matrix operand order checked by reading, not run on hardware
//
#include <hip/hip_runtime.h>
#include <hip/hip_bf16.h>
#include <math.h>

// ---------------- problem constants (match reference) ----------------
#define BATCH   64
#define DIM     128
#define NPER    512
#define NNODES  (BATCH * NPER)          // 32768
#define TEMP_INV 10.0f                  // 1/TEMP
#define REG_INV  10.0f                  // 1/REG  (P0 = exp(dot/REG) = exp(10*dot))
#define MAXIT   100
#define AVAL    (1.0f / 512.0f)

// tiles: 16x16 output tiles, 32x32 tiles per batch, 1024 tiles/batch, 65536 total
#define WAVES_PER_WG    8
#define GEMM_WGS        (BATCH * 32 * 4)              // 8192 (b x tn x tmGroup)

typedef __attribute__((ext_vector_type(16))) __bf16 v16bf;
typedef __attribute__((ext_vector_type(8)))  float  v8f;

// ---------------- helpers ----------------
__device__ __forceinline__ unsigned short f2bf(float f) {
    unsigned int u = __float_as_uint(f);
    u += 0x7fffu + ((u >> 16) & 1u);   // round-to-nearest-even
    return (unsigned short)(u >> 16);
}
__device__ __forceinline__ float bf_lo(unsigned int u) {   // low bf16 of dword
    return __uint_as_float(u << 16);
}
__device__ __forceinline__ float bf_hi(unsigned int u) {   // high bf16 of dword
    return __uint_as_float(u & 0xffff0000u);
}

union FragU { uint4 u4[2]; v16bf v; };

// Load a 16(M) x 32(K) bf16 fragment from row-major [*, DIM] bf16 matrix.
// CDNA5 16-bit A layout: lanes 0-15 hold row M=lane, K = {k0..k0+7, k0+16..k0+23};
// lanes 16-31 hold row M=lane-16, K = {k0+8..k0+15, k0+24..k0+31}.
// B (32x16) uses the identical striping on rows of st (B = st^T).
__device__ __forceinline__ v16bf load_frag(const unsigned short* base, int row0, int k0, int lane) {
    int r    = lane & 15;
    int half = lane >> 4;
    const unsigned short* p = base + (size_t)(row0 + r) * DIM + k0 + half * 8;
    FragU f;
    f.u4[0] = *(const uint4*)(p);        // K = k0+half*8 .. +7
    f.u4[1] = *(const uint4*)(p + 16);   // K = k0+16+half*8 .. +7
    return f.v;
}

// ---------------- kernel 1: row-normalize nodes/tokens -> bf16 ----------------
__global__ void k_normalize(const float* __restrict__ sn, const float* __restrict__ st,
                            unsigned short* __restrict__ snb, unsigned short* __restrict__ stb) {
    int wave = blockIdx.x * WAVES_PER_WG + (threadIdx.x >> 5);
    int lane = threadIdx.x & 31;
    const float* src; unsigned short* dst; int row;
    if (wave < NNODES) { src = sn; dst = snb; row = wave; }
    else               { src = st; dst = stb; row = wave - NNODES; }
    float4 x = *(const float4*)(src + (size_t)row * DIM + lane * 4);
    float ss = x.x * x.x + x.y * x.y + x.z * x.z + x.w * x.w;
    #pragma unroll
    for (int o = 16; o; o >>= 1) ss += __shfl_xor(ss, o, 32);
    float scale = 1.0f / fmaxf(sqrtf(ss), 1e-12f);
    unsigned short* d = dst + (size_t)row * DIM + lane * 4;
    ushort4 out;
    out.x = f2bf(x.x * scale); out.y = f2bf(x.y * scale);
    out.z = f2bf(x.z * scale); out.w = f2bf(x.w * scale);
    *(ushort4*)d = out;
}

// ---------------- kernel 2: global NT-Xent (single workgroup, 128 threads) ----
__global__ void k_global_loss(const float* __restrict__ sg, const float* __restrict__ qg,
                              float* __restrict__ out) {
    __shared__ float invS[BATCH], invQ[BATCH];
    __shared__ float sim[BATCH * BATCH];
    __shared__ float red[128];
    int tid = threadIdx.x;
    {   // per-row inverse norms
        int r = tid & 63;
        const float* src = (tid < 64) ? sg : qg;
        float ss = 0.f;
        for (int k = 0; k < DIM; ++k) { float x = src[(size_t)r * DIM + k]; ss += x * x; }
        float inv = 1.0f / fmaxf(sqrtf(ss), 1e-12f);
        if (tid < 64) invS[r] = inv; else invQ[r] = inv;
    }
    __syncthreads();
    for (int idx = tid; idx < BATCH * BATCH; idx += 128) {
        int i = idx >> 6, j = idx & 63;
        float d = 0.f;
        for (int k = 0; k < DIM; ++k) d += sg[(size_t)i * DIM + k] * qg[(size_t)j * DIM + k];
        sim[idx] = d * invS[i] * invQ[j] * TEMP_INV;
    }
    __syncthreads();
    {   // threads 0..63: row log-softmax diag; 64..127: column
        int r = tid & 63;
        bool rowDir = (tid < 64);
        float mx = -INFINITY;
        for (int j = 0; j < 64; ++j) {
            float s = rowDir ? sim[r * 64 + j] : sim[j * 64 + r];
            mx = fmaxf(mx, s);
        }
        float se = 0.f;
        for (int j = 0; j < 64; ++j) {
            float s = rowDir ? sim[r * 64 + j] : sim[j * 64 + r];
            se += __expf(s - mx);
        }
        float lse = mx + __logf(se);
        red[tid] = sim[r * 64 + r] - lse;
    }
    __syncthreads();
    if (tid == 0) {
        float s = 0.f;
        for (int i = 0; i < 128; ++i) s += red[i];
        out[0] = -s / 128.0f;   // (loss_i + loss_j)/2 with means over 64
    }
}

// ---------------- kernel 3: WMMA GEMM -> P0 (bf16) and P0^T (bf16) ------------
// WG covers a 128(m) x 16(n) output slab: wave w handles tm = tmg*8+w, tn fixed.
// P0 stores are 32B-coalesced directly; P0^T goes through an LDS transpose so
// the global stores become 16 rows x 256 contiguous bytes (b128 per thread).
__global__ void k_gemm_p0(const unsigned short* __restrict__ snb,
                          const unsigned short* __restrict__ stb,
                          unsigned short* __restrict__ P0,
                          unsigned short* __restrict__ P0T) {
    __shared__ unsigned short tr[16][144];   // row stride 288B (16B multiple)
    int wid  = threadIdx.x >> 5;
    int lane = threadIdx.x & 31;
    int wg   = blockIdx.x;        // 8192
    int b    = wg >> 7;           // / 128
    int rem  = wg & 127;
    int tn   = rem >> 2;          // 0..31
    int tmg  = rem & 3;           // 0..3
    int tm   = tmg * 8 + wid;     // 0..31
    const unsigned short* A  = snb + (size_t)b * NPER * DIM;
    const unsigned short* Bm = stb + (size_t)b * NPER * DIM;

    v8f c = {};
    #pragma unroll
    for (int k0 = 0; k0 < DIM; k0 += 32) {
        v16bf a  = load_frag(A,  tm * 16, k0, lane);
        v16bf bb = load_frag(Bm, tn * 16, k0, lane);
        c = __builtin_amdgcn_wmma_f32_16x16x32_bf16(false, a, false, bb,
                                                    (short)0, c, false, false);
    }
    int n  = lane & 15;
    int mh = (lane >> 4) * 8;
    unsigned short* p0b = P0 + (size_t)b * NPER * NPER;
    #pragma unroll
    for (int i = 0; i < 8; ++i) {
        int m = mh + i;
        float dot = c[i];                       // cost C = -dot
        float p   = __expf(dot * REG_INV);      // P0 = exp(-C/REG) = exp(10*dot)
        unsigned short pb = f2bf(p);
        int rg = tm * 16 + m, cg = tn * 16 + n;
        p0b[(size_t)rg * NPER + cg] = pb;       // 16 lanes -> 32B contiguous
        tr[n][wid * 16 + m] = pb;               // stage transpose in LDS
    }
    __syncthreads();
    // coalesced P0^T store: 16 rows x 128 bf16; 256 threads x 8 bf16 (b128)
    int rr = threadIdx.x >> 4;   // 0..15 : row n within tn tile
    int ch = threadIdx.x & 15;   // 0..15 : 8-element chunk along m
    uint4 val = *(const uint4*)&tr[rr][ch * 8];
    unsigned short* p0t = P0T + (size_t)b * NPER * NPER
                        + (size_t)(tn * 16 + rr) * NPER + tmg * 128 + ch * 8;
    *(uint4*)p0t = val;
}

// ---------------- kernel 4: init u, v ----------------
__global__ void k_init_uv(float* __restrict__ u, float* __restrict__ v) {
    int i = blockIdx.x * blockDim.x + threadIdx.x;
    if (i < BATCH * NPER) { u[i] = AVAL; v[i] = AVAL; }
}

// ---------------- kernel 5a: Sinkhorn partial sums (K split in half) ----------
// grid = BATCH*2 (b x kh), 128 threads; thread owns 4 consecutive columns and
// streams b64 (4xbf16) loads from P0 and P0^T rows k in [kh*256, kh*256+256).
__global__ void k_sinkhorn_part(const unsigned short* __restrict__ P0,
                                const unsigned short* __restrict__ P0T,
                                const float* __restrict__ u_in, const float* __restrict__ v_in,
                                float* __restrict__ pr, float* __restrict__ pc) {
    __shared__ float uL[256], vL[256];
    int b = blockIdx.x >> 1, kh = blockIdx.x & 1;
    for (int i = threadIdx.x; i < 256; i += blockDim.x) {
        uL[i] = u_in[b * NPER + kh * 256 + i];
        vL[i] = v_in[b * NPER + kh * 256 + i];
    }
    __syncthreads();
    int t0 = threadIdx.x * 4;                       // 4 columns per thread
    const unsigned short* p0  = P0  + (size_t)b * NPER * NPER + (size_t)kh * 256 * NPER + t0;
    const unsigned short* p0t = P0T + (size_t)b * NPER * NPER + (size_t)kh * 256 * NPER + t0;
    float4 ar = {0.f, 0.f, 0.f, 0.f};
    float4 ac = {0.f, 0.f, 0.f, 0.f};
    #pragma unroll 4
    for (int k = 0; k < 256; ++k) {
        uint2 rt = *(const uint2*)(p0t + (size_t)k * NPER);  // 4 bf16 of P0^T row
        uint2 rp = *(const uint2*)(p0  + (size_t)k * NPER);  // 4 bf16 of P0 row
        float vk = vL[k], uk = uL[k];
        ar.x += bf_lo(rt.x) * vk; ar.y += bf_hi(rt.x) * vk;
        ar.z += bf_lo(rt.y) * vk; ar.w += bf_hi(rt.y) * vk;
        ac.x += bf_lo(rp.x) * uk; ac.y += bf_hi(rp.x) * uk;
        ac.z += bf_lo(rp.y) * uk; ac.w += bf_hi(rp.y) * uk;
    }
    // partial layout: [kh][b][512]
    size_t o = ((size_t)kh * BATCH + b) * NPER + t0;
    *(float4*)(pr + o) = ar;
    *(float4*)(pc + o) = ac;
}

// ---------------- kernel 5b: combine partials -> u_new, v_new -----------------
__global__ void k_sinkhorn_comb(const float* __restrict__ pr, const float* __restrict__ pc,
                                float* __restrict__ u_out, float* __restrict__ v_out) {
    int b = blockIdx.x;
    int t0 = threadIdx.x * 4;
    size_t o0 = (size_t)b * NPER + t0;
    size_t o1 = ((size_t)BATCH + b) * NPER + t0;
    float4 r0 = *(const float4*)(pr + o0), r1 = *(const float4*)(pr + o1);
    float4 c0 = *(const float4*)(pc + o0), c1 = *(const float4*)(pc + o1);
    float4 un, vn;
    un.x = AVAL / (r0.x + r1.x); un.y = AVAL / (r0.y + r1.y);
    un.z = AVAL / (r0.z + r1.z); un.w = AVAL / (r0.w + r1.w);
    vn.x = AVAL / (c0.x + c1.x); vn.y = AVAL / (c0.y + c1.y);
    vn.z = AVAL / (c0.z + c1.z); vn.w = AVAL / (c0.w + c1.w);
    // nan_to_num(nan/±inf -> 1/n)
    if (!__builtin_isfinite(un.x)) un.x = AVAL;
    if (!__builtin_isfinite(un.y)) un.y = AVAL;
    if (!__builtin_isfinite(un.z)) un.z = AVAL;
    if (!__builtin_isfinite(un.w)) un.w = AVAL;
    if (!__builtin_isfinite(vn.x)) vn.x = AVAL;
    if (!__builtin_isfinite(vn.y)) vn.y = AVAL;
    if (!__builtin_isfinite(vn.z)) vn.z = AVAL;
    if (!__builtin_isfinite(vn.w)) vn.w = AVAL;
    *(float4*)(u_out + (size_t)b * NPER + t0) = un;
    *(float4*)(v_out + (size_t)b * NPER + t0) = vn;
}

// ---------------- kernel 6: sum(P * C) via WMMA recompute, per-WG partials ----
__global__ void k_reduce_tiles(const unsigned short* __restrict__ snb,
                               const unsigned short* __restrict__ stb,
                               const float* __restrict__ u, const float* __restrict__ v,
                               float* __restrict__ partials) {
    __shared__ float wsum[WAVES_PER_WG];
    int wid  = threadIdx.x >> 5;
    int lane = threadIdx.x & 31;
    int wave = blockIdx.x * WAVES_PER_WG + wid;
    int b  = wave >> 10;
    int t  = wave & 1023;
    int tm = t >> 5, tn = t & 31;
    const unsigned short* A  = snb + (size_t)b * NPER * DIM;
    const unsigned short* Bm = stb + (size_t)b * NPER * DIM;

    v8f c = {};
    #pragma unroll
    for (int k0 = 0; k0 < DIM; k0 += 32) {
        v16bf a  = load_frag(A,  tm * 16, k0, lane);
        v16bf bb = load_frag(Bm, tn * 16, k0, lane);
        c = __builtin_amdgcn_wmma_f32_16x16x32_bf16(false, a, false, bb,
                                                    (short)0, c, false, false);
    }
    int n  = lane & 15;
    int mh = (lane >> 4) * 8;
    const float* ub = u + b * NPER;
    const float* vb = v + b * NPER;
    float vn = vb[tn * 16 + n];
    float acc = 0.f;
    #pragma unroll
    for (int i = 0; i < 8; ++i) {
        int m = mh + i;
        float dot = c[i];
        float p0  = __expf(dot * REG_INV);
        // P*C = u * P0 * v * (-dot)
        acc += ub[tm * 16 + m] * p0 * vn * (-dot);
    }
    #pragma unroll
    for (int o = 16; o; o >>= 1) acc += __shfl_xor(acc, o, 32);
    if (lane == 0) wsum[wid] = acc;
    __syncthreads();
    if (threadIdx.x == 0) {
        float s = 0.f;
        #pragma unroll
        for (int i = 0; i < WAVES_PER_WG; ++i) s += wsum[i];
        partials[blockIdx.x] = s;
    }
}

// ---------------- kernel 7: deterministic final reduction ----------------
__global__ void k_finalize(const float* __restrict__ partials, float* __restrict__ out) {
    __shared__ float red[256];
    int tid = threadIdx.x;
    float a = 0.f;
    for (int i = tid; i < GEMM_WGS; i += 256) a += partials[i];
    red[tid] = a;
    __syncthreads();
    #pragma unroll
    for (int s = 128; s; s >>= 1) {
        if (tid < s) red[tid] += red[tid + s];
        __syncthreads();
    }
    if (tid == 0) {
        float local = red[0] / (float)BATCH;
        out[1] = local;
        out[2] = out[0] + local;   // out[0] already written by k_global_loss
    }
}

// ---------------- host launcher ----------------
extern "C" void kernel_launch(void* const* d_in, const int* in_sizes, int n_in,
                              void* d_out, int out_size, void* d_ws, size_t ws_size,
                              hipStream_t stream) {
    const float* struct_global = (const float*)d_in[0];   // [64,128]
    const float* seq_global    = (const float*)d_in[1];   // [64,128]
    const float* struct_nodes  = (const float*)d_in[2];   // [32768,128]
    const float* seq_tokens    = (const float*)d_in[3];   // [32768,128]
    float* out = (float*)d_out;                           // 3 floats

    // workspace layout
    char* w = (char*)d_ws;
    size_t off = 0;
    unsigned short* snb = (unsigned short*)(w + off); off += (size_t)NNODES * DIM * 2;          // 8 MB
    unsigned short* stb = (unsigned short*)(w + off); off += (size_t)NNODES * DIM * 2;          // 8 MB
    unsigned short* P0  = (unsigned short*)(w + off); off += (size_t)BATCH * NPER * NPER * 2;   // 32 MB
    unsigned short* P0T = (unsigned short*)(w + off); off += (size_t)BATCH * NPER * NPER * 2;   // 32 MB
    float* u0 = (float*)(w + off); off += (size_t)BATCH * NPER * 4;
    float* v0 = (float*)(w + off); off += (size_t)BATCH * NPER * 4;
    float* u1 = (float*)(w + off); off += (size_t)BATCH * NPER * 4;
    float* v1 = (float*)(w + off); off += (size_t)BATCH * NPER * 4;
    float* pr = (float*)(w + off); off += (size_t)2 * BATCH * NPER * 4;  // K-split partials
    float* pc = (float*)(w + off); off += (size_t)2 * BATCH * NPER * 4;
    float* partials = (float*)(w + off); off += (size_t)GEMM_WGS * 4;
    (void)off; (void)ws_size; (void)in_sizes; (void)n_in; (void)out_size;

    // 1) normalize nodes/tokens to bf16 (one wave per row, 65536 waves)
    k_normalize<<<dim3(GEMM_WGS), dim3(256), 0, stream>>>(struct_nodes, seq_tokens, snb, stb);

    // 2) global NT-Xent loss -> out[0]
    k_global_loss<<<dim3(1), dim3(128), 0, stream>>>(struct_global, seq_global, out);

    // 3) batched WMMA GEMM: P0 = exp(10 * sn.st^T), plus LDS-transposed P0^T
    k_gemm_p0<<<dim3(GEMM_WGS), dim3(256), 0, stream>>>(snb, stb, P0, P0T);

    // 4) init u, v = 1/512
    k_init_uv<<<dim3((BATCH * NPER + 255) / 256), dim3(256), 0, stream>>>(u0, v0);

    // 5) 100 Jacobi Sinkhorn iterations (ping-pong u/v; grid sync = launch boundary)
    float* ub[2] = { u0, u1 };
    float* vb[2] = { v0, v1 };
    for (int it = 0; it < MAXIT; ++it) {
        int in = it & 1, on = in ^ 1;
        k_sinkhorn_part<<<dim3(BATCH * 2), dim3(128), 0, stream>>>(P0, P0T, ub[in], vb[in], pr, pc);
        k_sinkhorn_comb<<<dim3(BATCH), dim3(128), 0, stream>>>(pr, pc, ub[on], vb[on]);
    }
    // after 100 iters, final (u,v) live in buffers index 0

    // 6) sum(P*C)/B with WMMA-recomputed C, deterministic two-stage reduction
    k_reduce_tiles<<<dim3(GEMM_WGS), dim3(256), 0, stream>>>(snb, stb, ub[0], vb[0], partials);
    k_finalize<<<dim3(1), dim3(256), 0, stream>>>(partials, out);
}
